// BinaryTreeLSTMCell_15960098472359
// MI455X (gfx1250) — compile-verified, matrix operand order checked by reading, not run on hardware
//
#include <hip/hip_runtime.h>
#include <hip/hip_bf16.h>
#include <stdint.h>

// ---------------------------------------------------------------------------
// BinaryTreeLSTMCell on MI455X (gfx1250)
//   gates(8192x5120) = concat(x,h_left,h_right)(8192x3072) @ W^T + b
// bf16 WMMA GEMM fed by Tensor Data Mover (TDM) global->LDS tiles,
// double-buffered with s_wait_tensorcnt, fully fused LSTM epilogue.
// Block tile 128(M) x 64(N) x 5 gates; wave tile 64(M) x 16(N) x 5 gates.
// ---------------------------------------------------------------------------

#define BATCH   8192
#define KDIM    3072
#define HIDDEN  1024
#define NGATES  5

#define BM      128         // block M tile
#define BN      64          // block N tile (per gate)
#define KSTEP   32
#define PITCH   40          // LDS row pitch in bf16 elems (32 data + 8 pad)

typedef __attribute__((ext_vector_type(16))) __bf16 v16bf;
typedef __attribute__((ext_vector_type(8)))  __bf16 v8bf;
typedef __attribute__((ext_vector_type(4)))  __bf16 v4bf;
typedef __attribute__((ext_vector_type(8)))  float  v8f;
typedef __attribute__((ext_vector_type(4)))  unsigned int v4u;
typedef __attribute__((ext_vector_type(8)))  int    v8i;
typedef __attribute__((ext_vector_type(4)))  int    v4i;

// ---------------- prep: fp32 -> bf16 packing ------------------------------

__global__ void pack_combined_bf16(const float* __restrict__ x,
                                   const float* __restrict__ hl,
                                   const float* __restrict__ hr,
                                   __bf16* __restrict__ out) {
    size_t base = ((size_t)blockIdx.x * blockDim.x + threadIdx.x) * 4;
    if (base >= (size_t)BATCH * KDIM) return;
    int row = (int)(base / KDIM);
    int k   = (int)(base % KDIM);
    const float* src;
    if (k < 1024)       src = x  + (size_t)row * 1024 + k;
    else if (k < 2048)  src = hl + (size_t)row * 1024 + (k - 1024);
    else                src = hr + (size_t)row * 1024 + (k - 2048);
    float4 v = *reinterpret_cast<const float4*>(src);
    v4bf o;
    o[0] = (__bf16)v.x; o[1] = (__bf16)v.y; o[2] = (__bf16)v.z; o[3] = (__bf16)v.w;
    *reinterpret_cast<v4bf*>(out + base) = o;
}

__global__ void pack_w_bf16(const float* __restrict__ w,
                            __bf16* __restrict__ out) {
    size_t base = ((size_t)blockIdx.x * blockDim.x + threadIdx.x) * 4;
    if (base >= (size_t)NGATES * HIDDEN * KDIM) return;
    float4 v = *reinterpret_cast<const float4*>(w + base);
    v4bf o;
    o[0] = (__bf16)v.x; o[1] = (__bf16)v.y; o[2] = (__bf16)v.z; o[3] = (__bf16)v.w;
    *reinterpret_cast<v4bf*>(out + base) = o;
}

// ---------------- TDM descriptors ------------------------------------------
// D# packing per CDNA5 ISA (CDNA5 ISA). Groups 1..3 are fully
// compile-time constant here; only group0 (lds addr + global addr) varies.
// Group1 word0: mask=0, data_size=1(2B), pad_enable=1, pad_interval=3(16 DW),
// pad_amount=3(4 DW) -> LDS pitch = 32+8 halves = 80 B (bank-conflict-free).
#define TDM_W0  (int)((1u << 16) | (1u << 20) | (3u << 22) | (3u << 25))

// A: 2D tile KSTEP x BM from tensor (BATCH x KDIM), stride0 = KDIM
#define G1_A (v8i){ TDM_W0,                                                  \
                    (int)((KDIM & 0xFFFF) << 16),                            \
                    (int)(((KDIM >> 16) & 0xFFFF) | ((BATCH & 0xFFFF) << 16)),\
                    (int)(((BATCH >> 16) & 0xFFFF) | (KSTEP << 16)),         \
                    (int)(BM),                                               \
                    (int)(KDIM), 0, 0 }

// B: 3D tile KSTEP x BN x NGATES from (NGATES*HIDDEN x KDIM),
// stride0 = KDIM, stride1 (gate) = HIDDEN*KDIM = 0x300000
#define G1_B (v8i){ TDM_W0,                                                  \
                    (int)((KDIM & 0xFFFF) << 16),                            \
                    (int)(((KDIM >> 16) & 0xFFFF) | ((HIDDEN & 0xFFFF) << 16)),\
                    (int)(((HIDDEN >> 16) & 0xFFFF) | (KSTEP << 16)),        \
                    (int)(BN | (NGATES << 16)),                              \
                    (int)(KDIM), 0,                                          \
                    (int)(((uint64_t)HIDDEN * KDIM) >> 16) }

__device__ __forceinline__ void tdm_issue(uint32_t lds_addr, uint64_t ga,
                                          v8i g1, v4i g2, v4i g3) {
    v4u g0 = { 1u,
               lds_addr,
               (uint32_t)ga,
               (uint32_t)(((ga >> 32) & 0x1FFFFFFu) | (2u << 30)) };
#if __has_include(<hip/amd_detail/amd_gfx1250_TDM.h>)
    v8i gz = { 0, 0, 0, 0, 0, 0, 0, 0 };
    __builtin_amdgcn_tensor_load_to_lds(g0, g1, g2, g3, gz, 0);
#else
    __builtin_amdgcn_tensor_load_to_lds(g0, g1, g2, g3, 0);
#endif
}

// ---------------- WMMA fragment from LDS -----------------------------------
// 16-bit A/B operand layout (ISA 7.12.2): lane L -> row L%16,
// K(j) = 8*(L/16) + (j<8 ? j : j+8). Two 16B ds loads per fragment.
__device__ __forceinline__ v16bf frag_from_lds(const __bf16* __restrict__ buf,
                                               int row0, int lane) {
    int r  = row0 + (lane & 15);
    int kb = (lane >> 4) << 3;
    const __bf16* p = buf + r * PITCH + kb;
    v8bf lo = *reinterpret_cast<const v8bf*>(p);
    v8bf hi = *reinterpret_cast<const v8bf*>(p + 16);
    v16bf f;
#pragma unroll
    for (int i = 0; i < 8; ++i) { f[i] = lo[i]; f[i + 8] = hi[i]; }
    return f;
}

__device__ __forceinline__ float fast_sigmoid(float x) {
    return 1.0f / (1.0f + __expf(-x));
}
__device__ __forceinline__ float fast_tanh(float x) {
    return 1.0f - 2.0f / (__expf(2.0f * x) + 1.0f);
}

// ---------------- fused GEMM + LSTM epilogue ------------------------------
// 256 threads = 8 wave32s arranged 2(M) x 4(N).
// Wave tile: 64(M) x 16(N) x 5 gates -> 20 v8f accumulators, 20 WMMA /K-step,
// fed by 18 ds_load_b128 (0.9 loads/WMMA).
__global__ void __launch_bounds__(256)
tree_lstm_wmma_kernel(const __bf16* __restrict__ A,     // 8192 x 3072
                      const __bf16* __restrict__ Wb,    // 5120 x 3072
                      const float*  __restrict__ bias,  // 5120
                      const float*  __restrict__ c_left,
                      const float*  __restrict__ c_right,
                      float* __restrict__ out_h,
                      float* __restrict__ out_c) {
    __shared__ __bf16 Abuf[2][BM * PITCH];           // 2 x 10.0 KB
    __shared__ __bf16 Bbuf[2][NGATES * BN * PITCH];  // 2 x 25.6 KB

    const int lane = threadIdx.x & 31;
    const int wid  = threadIdx.x >> 5;
    const int wm   = wid & 1;          // 0..1 : 64-row slab
    const int wn   = wid >> 1;         // 0..3 : 16-col slab

    const int m0blk = blockIdx.x * BM;
    const int n0blk = blockIdx.y * BN;

    const uint64_t gaA0 = (uint64_t)(uintptr_t)(A  + (size_t)m0blk * KDIM);
    const uint64_t gaB0 = (uint64_t)(uintptr_t)(Wb + (size_t)n0blk * KDIM);

    const v4i g2A = { 0, 0, 0, 0 };
    const v4i g2B = { NGATES, 0, 0, 0 };
    const v4i g3z = { 0, 0, 0, 0 };

    const int NITER = KDIM / KSTEP;    // 96

    if (wid == 0) {   // prologue: fill buffer 0
        tdm_issue((uint32_t)(uintptr_t)&Abuf[0][0], gaA0, G1_A, g2A, g3z);
        tdm_issue((uint32_t)(uintptr_t)&Bbuf[0][0], gaB0, G1_B, g2B, g3z);
    }

    v8f acc[NGATES][4] = {};

    for (int it = 0; it < NITER; ++it) {
        const int cur = it & 1;
        if (wid == 0 && it + 1 < NITER) {   // prefetch next K-slice
            const uint64_t koff = (uint64_t)(it + 1) * (KSTEP * 2);  // bytes
            tdm_issue((uint32_t)(uintptr_t)&Abuf[cur ^ 1][0], gaA0 + koff,
                      G1_A, g2A, g3z);
            tdm_issue((uint32_t)(uintptr_t)&Bbuf[cur ^ 1][0], gaB0 + koff,
                      G1_B, g2B, g3z);
        }
        if (wid == 0) {
            if (it + 1 < NITER) __builtin_amdgcn_s_wait_tensorcnt(2);
            else                __builtin_amdgcn_s_wait_tensorcnt(0);
        }
        __syncthreads();   // current buffer visible to all waves

        const __bf16* Ab = &Abuf[cur][0];
        const __bf16* Bb = &Bbuf[cur][0];

        v16bf a0 = frag_from_lds(Ab, wm * 64 + 0, lane);
        v16bf a1 = frag_from_lds(Ab, wm * 64 + 16, lane);
        v16bf a2 = frag_from_lds(Ab, wm * 64 + 32, lane);
        v16bf a3 = frag_from_lds(Ab, wm * 64 + 48, lane);
#pragma unroll
        for (int g = 0; g < NGATES; ++g) {
            v16bf bf = frag_from_lds(Bb, g * BN + wn * 16, lane);
            acc[g][0] = __builtin_amdgcn_wmma_f32_16x16x32_bf16(
                false, a0, false, bf, (short)0, acc[g][0], false, false);
            acc[g][1] = __builtin_amdgcn_wmma_f32_16x16x32_bf16(
                false, a1, false, bf, (short)0, acc[g][1], false, false);
            acc[g][2] = __builtin_amdgcn_wmma_f32_16x16x32_bf16(
                false, a2, false, bf, (short)0, acc[g][2], false, false);
            acc[g][3] = __builtin_amdgcn_wmma_f32_16x16x32_bf16(
                false, a3, false, bf, (short)0, acc[g][3], false, false);
        }
        __syncthreads();   // all reads done before buffer is overwritten
    }

    // Epilogue. C/D layout: VGPR r, lanes 0-15 -> M=r,   N=lane;
    //                               lanes 16-31 -> M=8+r, N=lane-16.
    const int nloc = lane & 15;
    const int mhi  = (lane >> 4) << 3;
    const int n    = n0blk + wn * 16 + nloc;

    const float bi = bias[0 * HIDDEN + n];
    const float bl = bias[1 * HIDDEN + n];
    const float br = bias[2 * HIDDEN + n];
    const float bo = bias[3 * HIDDEN + n];
    const float bu = bias[4 * HIDDEN + n];

#pragma unroll
    for (int t = 0; t < 4; ++t) {
#pragma unroll
        for (int r = 0; r < 8; ++r) {
            const int m = m0blk + wm * 64 + t * 16 + mhi + r;
            const size_t idx = (size_t)m * HIDDEN + n;

            const float gi = fast_sigmoid(acc[0][t][r] + bi);
            const float gl = fast_sigmoid(acc[1][t][r] + bl);
            const float gr = fast_sigmoid(acc[2][t][r] + br);
            const float go = fast_sigmoid(acc[3][t][r] + bo);
            const float gu = fast_tanh   (acc[4][t][r] + bu);

            const float cl = c_left[idx];
            const float cr = c_right[idx];

            const float c = gi * gu + gl * cl + gr * cr;
            const float h = go * fast_tanh(c);

            out_h[idx] = h;
            out_c[idx] = c;
        }
    }
}

// ---------------- launch ---------------------------------------------------

extern "C" void kernel_launch(void* const* d_in, const int* in_sizes, int n_in,
                              void* d_out, int out_size, void* d_ws, size_t ws_size,
                              hipStream_t stream) {
    const float* x       = (const float*)d_in[0];
    const float* h_left  = (const float*)d_in[1];
    const float* c_left  = (const float*)d_in[2];
    const float* h_right = (const float*)d_in[3];
    const float* c_right = (const float*)d_in[4];
    const float* W       = (const float*)d_in[5];
    const float* b       = (const float*)d_in[6];

    __bf16* combined_bf = (__bf16*)d_ws;
    __bf16* w_bf        = combined_bf + (size_t)BATCH * KDIM;

    {
        size_t threads = (size_t)BATCH * KDIM / 4;
        int blocks = (int)((threads + 255) / 256);
        pack_combined_bf16<<<blocks, 256, 0, stream>>>(x, h_left, h_right, combined_bf);
    }
    {
        size_t threads = (size_t)NGATES * HIDDEN * KDIM / 4;
        int blocks = (int)((threads + 255) / 256);
        pack_w_bf16<<<blocks, 256, 0, stream>>>(W, w_bf);
    }

    float* out_h = (float*)d_out;
    float* out_c = out_h + (size_t)BATCH * HIDDEN;

    dim3 grid(BATCH / BM, HIDDEN / BN);
    tree_lstm_wmma_kernel<<<grid, 256, 0, stream>>>(
        combined_bf, w_bf, b, c_left, c_right, out_h, out_c);
}
